// RNNGraph_61340722921888
// MI455X (gfx1250) — matrix operand order repaired and checked by reading, not compiled, not run on hardware
//
#include <hip/hip_runtime.h>
#include <hip/hip_bf16.h>
#include <cstdint>

// ---- problem constants (match reference) ----
#define TT   65536
#define CC   8
#define FF   10
#define HH   64
#define G3   192    // 3*H
#define NIN  4

typedef __attribute__((ext_vector_type(16))) _Float16 v16h;
typedef __attribute__((ext_vector_type(8)))  _Float16 v8h;
typedef __attribute__((ext_vector_type(8)))  float    v8f;

// Single v_rcp_f32 instead of IEEE divide chain (gate math tolerates 1ulp rcp)
__device__ __forceinline__ float fast_rcp(float x) {
#if __has_builtin(__builtin_amdgcn_rcpf)
    return __builtin_amdgcn_rcpf(x);
#else
    return 1.0f / x;
#endif
}

__device__ __forceinline__ float fast_sigmoid(float x) {
    return fast_rcp(1.0f + __expf(-x));
}

// Hardware v_tanh_f32 when available; else exp2+rcp formulation (no libm blowup)
__device__ __forceinline__ float fast_tanh(float x) {
#if __has_builtin(__builtin_amdgcn_tanhf)
    return __builtin_amdgcn_tanhf(x);
#else
    // tanh(x) = 1 - 2/(exp(2x)+1)
    float e = __expf(2.0f * x);
    return 1.0f - 2.0f * fast_rcp(e + 1.0f);
#endif
}

// ---------------------------------------------------------------------------
// Kernel 0: init y[t] = sum_c mask ? b_out[c] : 0 ; zero count/cursor
// ---------------------------------------------------------------------------
__global__ void init_kernel(const float* __restrict__ b_out,
                            const int* __restrict__ masks,
                            float* __restrict__ y,
                            int* __restrict__ counters /* count[C] then cursor[C] */) {
    int t = blockIdx.x * blockDim.x + threadIdx.x;
    if (blockIdx.x == 0 && threadIdx.x < 2 * CC) counters[threadIdx.x] = 0;
    if (t < TT) {
        float acc = 0.f;
        #pragma unroll
        for (int c = 0; c < CC; ++c)
            if (masks[(size_t)t * CC + c] == 1) acc += b_out[c];   // NOUT=1
        y[t] = acc;
    }
}

// ---------------------------------------------------------------------------
// Kernel 1: gx[t][c][g] = sum_f W_ih[c][g][f]*x[t][1+f][c] + b_ih + (g<128? b_hh : 0)
// (b_hh for r/z gates folded in; n-gate b_hh goes into the WMMA accumulator)
// ---------------------------------------------------------------------------
__global__ void gx_kernel(const float* __restrict__ x,
                          const float* __restrict__ W_ih,
                          const float* __restrict__ b_ih,
                          const float* __restrict__ b_hh,
                          float* __restrict__ gx) {
    size_t idx = (size_t)blockIdx.x * blockDim.x + threadIdx.x;
    if (idx >= (size_t)TT * CC * G3) return;
    int g = (int)(idx % G3);
    int c = (int)((idx / G3) % CC);
    size_t t = idx / ((size_t)G3 * CC);
    const float* w  = W_ih + ((size_t)c * G3 + g) * NIN;
    const float* xp = x + (t * FF + 1) * CC + c;
    float acc = b_ih[c * G3 + g] + (g < 128 ? b_hh[c * G3 + g] : 0.0f);
    #pragma unroll
    for (int f = 0; f < NIN; ++f) acc += w[f] * xp[(size_t)f * CC];
    gx[idx] = acc;
}

// ---------------------------------------------------------------------------
// Kernel 2: discover segment starts per channel (order irrelevant: independent)
// ---------------------------------------------------------------------------
__global__ void seg_kernel(const int* __restrict__ starts,
                           int* __restrict__ count,
                           int* __restrict__ seg_start) {
    size_t idx = (size_t)blockIdx.x * blockDim.x + threadIdx.x;
    if (idx >= (size_t)TT * CC) return;
    int c = (int)(idx % CC);
    int t = (int)(idx / CC);
    if (starts[idx] == 1) {
        int p = atomicAdd(&count[c], 1);
        seg_start[(size_t)c * TT + p] = t;
    }
}

// ---------------------------------------------------------------------------
// Kernel 3: segment-parallel GRU scan. One wave32 per block; 16 independent
// segments form the N dimension of v_wmma_f32_16x16x32_f16.
//   gh(192x16) = W_hh(192x64,f16) @ H(64x16,f16)  => 12 Mtiles x 2 Ktiles WMMA
// C/D layout: lane l, vgpr v holds (M = 16*mt + 8*(l>>4) + v, N = l&15)
// A layout  : lane l holds row M = l&15, K = (l>>4)*16 .. +15 (pre-packed in LDS)
// B layout  : lane l holds (N = l&15), K = (l>>4)*16 .. +15  => contiguous in h_lds
// ---------------------------------------------------------------------------
__global__ __launch_bounds__(32)
void scan_kernel(const float* __restrict__ W_hh,
                 const float* __restrict__ b_hh,
                 const float* __restrict__ W_out,
                 const int*   __restrict__ starts,
                 const int*   __restrict__ masks,
                 const float* __restrict__ gx,
                 const int*   __restrict__ seg_start,
                 const int*   __restrict__ count,
                 int*         __restrict__ cursor,
                 float*       __restrict__ y) {
    __shared__ __align__(64) _Float16 a_lds[24][512];   // 12 Mtiles x 2 Ktiles, swizzled A
    __shared__ __align__(64) _Float16 h_lds[16 * HH];   // h, natural [seg][j] f16
    __shared__ int s_active;
    __shared__ int s_batch;

    const int lane = threadIdx.x;       // 0..31
    const int c    = blockIdx.x % CC;
    const int n    = lane & 15;         // segment slot (N column)
    const int hi   = lane >> 4;

    // ---- one-time: W_hh (f32 row-major) -> LDS, pre-swizzled A-tile layout ----
    for (int mt = 0; mt < 12; ++mt) {
        for (int kt = 0; kt < 2; ++kt) {
            const int g = 16 * mt + n;                       // A row M = lane&15
            const float* wrow = W_hh + ((size_t)c * G3 + g) * HH + kt * 32 + hi * 16;
            v16h pk;
            #pragma unroll
            for (int i = 0; i < 8; ++i) pk[i]     = (_Float16)wrow[i];       // K = kt*32+hi*16+i
            #pragma unroll
            for (int i = 0; i < 8; ++i) pk[8 + i] = (_Float16)wrow[8 + i];   // K = +8
            *(v16h*)&a_lds[mt * 2 + kt][lane * 16] = pk;
        }
    }
    __syncthreads();

    // n-gate b_hh tiles in C/D layout (accumulator initializer)
    v8f bhh[4];
    #pragma unroll
    for (int mt = 0; mt < 4; ++mt)
        bhh[mt] = *(const v8f*)(b_hh + (size_t)c * G3 + 128 + 16 * mt + 8 * hi);

    // W_out rows this lane owns (rows 16*mt + 8*hi + v), NOUT==1
    float wout[32];
    #pragma unroll
    for (int mt = 0; mt < 4; ++mt)
        #pragma unroll
        for (int v = 0; v < 8; ++v)
            wout[mt * 8 + v] = W_out[(size_t)c * HH + 16 * mt + 8 * hi + v];

    const int nseg = count[c];

    for (;;) {
        if (lane == 0) s_batch = atomicAdd(&cursor[c], 16);
        __syncthreads();
        const int batch = s_batch;
        __syncthreads();
        if (batch >= nseg) break;

        const bool have = (batch + n) < nseg;
        const int  s    = have ? seg_start[(size_t)c * TT + (batch + n)] : 0;
        int  t      = s;
        bool active = have;

        int nact = nseg - batch; if (nact > 16) nact = 16;
        if (lane == 0) s_active = 2 * nact;                 // 2 lanes per segment
        for (int i = lane; i < 16 * HH; i += 32) h_lds[i] = (_Float16)0.0f;
        float h_old[32];
        #pragma unroll
        for (int i = 0; i < 32; ++i) h_old[i] = 0.0f;
        __syncthreads();

        for (;;) {
            if (s_active == 0) break;

            // B tiles (64x16 as two 32x16): contiguous 32B per lane
            const v16h B0 = *(const v16h*)&h_lds[n * HH + hi * 16];
            const v16h B1 = *(const v16h*)&h_lds[n * HH + 32 + hi * 16];

            const float* gxb = gx + ((size_t)t * CC + c) * G3;

            // speculative prefetch of gx row for channel c+1 at this t (768B = 3 lines)
            {
                const char* nxt = (const char*)(gxb + G3);
                __builtin_prefetch(nxt, 0, 3);
                __builtin_prefetch(nxt + 256, 0, 3);
                __builtin_prefetch(nxt + 512, 0, 3);
            }

            // r-gate tiles (gx+b_ih+b_hh folded into accumulator init)
            v8f racc[4];
            #pragma unroll
            for (int mt = 0; mt < 4; ++mt) {
                v8f acc = *(const v8f*)(gxb + 16 * mt + 8 * hi);
                v16h A0 = *(const v16h*)&a_lds[mt * 2 + 0][lane * 16];
                v16h A1 = *(const v16h*)&a_lds[mt * 2 + 1][lane * 16];
                acc = __builtin_amdgcn_wmma_f32_16x16x32_f16(false, A0, false, B0,
                                                             (short)0, acc, false, false);
                acc = __builtin_amdgcn_wmma_f32_16x16x32_f16(false, A1, false, B1,
                                                             (short)0, acc, false, false);
                racc[mt] = acc;
            }

            float partial = 0.0f;
            #pragma unroll
            for (int mt = 0; mt < 4; ++mt) {
                // z-gate: gx folded in
                v8f zacc = *(const v8f*)(gxb + 64 + 16 * mt + 8 * hi);
                {
                    v16h A0 = *(const v16h*)&a_lds[(4 + mt) * 2 + 0][lane * 16];
                    v16h A1 = *(const v16h*)&a_lds[(4 + mt) * 2 + 1][lane * 16];
                    zacc = __builtin_amdgcn_wmma_f32_16x16x32_f16(false, A0, false, B0,
                                                                  (short)0, zacc, false, false);
                    zacc = __builtin_amdgcn_wmma_f32_16x16x32_f16(false, A1, false, B1,
                                                                  (short)0, zacc, false, false);
                }
                // n-gate: gh kept separate (r multiplies gh only); acc init = b_hh
                v8f nacc = bhh[mt];
                {
                    v16h A0 = *(const v16h*)&a_lds[(8 + mt) * 2 + 0][lane * 16];
                    v16h A1 = *(const v16h*)&a_lds[(8 + mt) * 2 + 1][lane * 16];
                    nacc = __builtin_amdgcn_wmma_f32_16x16x32_f16(false, A0, false, B0,
                                                                  (short)0, nacc, false, false);
                    nacc = __builtin_amdgcn_wmma_f32_16x16x32_f16(false, A1, false, B1,
                                                                  (short)0, nacc, false, false);
                }
                v8f gxn = *(const v8f*)(gxb + 128 + 16 * mt + 8 * hi);
                #pragma unroll
                for (int v = 0; v < 8; ++v) {
                    float rr = fast_sigmoid(racc[mt][v]);
                    float zz = fast_sigmoid(zacc[v]);
                    float nn = fast_tanh(gxn[v] + rr * nacc[v]);
                    float hn = (1.0f - zz) * nn + zz * h_old[mt * 8 + v];
                    h_old[mt * 8 + v] = hn;
                    partial += wout[mt * 8 + v] * hn;
                }
            }

            // write h back to LDS in B-natural layout (8 f16 = one b128 per Mtile)
            #pragma unroll
            for (int mt = 0; mt < 4; ++mt) {
                v8h pk;
                #pragma unroll
                for (int v = 0; v < 8; ++v) pk[v] = (_Float16)h_old[mt * 8 + v];
                *(v8h*)&h_lds[n * HH + 16 * mt + 8 * hi] = pk;
            }
            asm volatile("s_wait_dscnt 0" ::: "memory");

            // masked per-(t,c) output contribution (both lanes hold half the dot)
            if (active && masks[(size_t)t * CC + c] == 1)
                atomicAdd(&y[t], partial);

            // advance segment time; deactivate at next contour start / end
            if (active) {
                int tn = t + 1;
                if (tn >= TT || starts[(size_t)tn * CC + c] == 1) {
                    active = false;
                    atomicSub(&s_active, 1);
                } else {
                    t = tn;
                }
            }
            __syncthreads();
        }
        __syncthreads();
    }
}

// ---------------------------------------------------------------------------
// launch
// ---------------------------------------------------------------------------
extern "C" void kernel_launch(void* const* d_in, const int* in_sizes, int n_in,
                              void* d_out, int out_size, void* d_ws, size_t ws_size,
                              hipStream_t stream) {
    const float* x      = (const float*)d_in[0];
    const int*   masks  = (const int*)  d_in[1];
    const int*   starts = (const int*)  d_in[2];
    const float* W_ih   = (const float*)d_in[3];
    const float* W_hh   = (const float*)d_in[4];
    const float* b_ih   = (const float*)d_in[5];
    const float* b_hh   = (const float*)d_in[6];
    const float* W_out  = (const float*)d_in[7];
    const float* b_out  = (const float*)d_in[8];
    float* y = (float*)d_out;

    // workspace layout
    int*   count     = (int*)d_ws;                        // [C]
    int*   cursor    = count + CC;                        // [C]
    int*   seg_start = cursor + CC;                       // [C*T]
    float* gx        = (float*)(seg_start + (size_t)CC * TT);  // [T*C*192]

    // 0) init output with masked b_out sum; zero counters
    init_kernel<<<(TT + 255) / 256, 256, 0, stream>>>(b_out, masks, y, count);

    // 1) precompute gx (input projection + folded biases)
    {
        size_t total = (size_t)TT * CC * G3;
        gx_kernel<<<(unsigned)((total + 255) / 256), 256, 0, stream>>>(x, W_ih, b_ih, b_hh, gx);
    }

    // 2) build per-channel segment lists
    seg_kernel<<<(TT * CC + 255) / 256, 256, 0, stream>>>(starts, count, seg_start);

    // 3) segment-parallel WMMA GRU scan (persistent wave32 blocks, 256/channel)
    scan_kernel<<<CC * 256, 32, 0, stream>>>(W_hh, b_hh, W_out, starts, masks,
                                             gx, seg_start, count, cursor, y);
}
